// TransformerBlock_47502338294550
// MI455X (gfx1250) — compile-verified, matrix operand order checked by reading, not run on hardware
//
#include <hip/hip_runtime.h>
#include <hip/hip_bf16.h>

// ---------------------------------------------------------------------------
// Types for CDNA5 WMMA (wave32, 16x16x32 bf16 -> f32)
// ---------------------------------------------------------------------------
typedef __bf16 bf16;
typedef bf16  v16bf __attribute__((ext_vector_type(16)));
typedef bf16  v8bf  __attribute__((ext_vector_type(8)));
typedef float v8f   __attribute__((ext_vector_type(8)));

static constexpr int D_MODEL = 1024;
static constexpr int N_HEADS = 16;
static constexpr int D_FF    = 4096;
static constexpr int SEQ_T   = 2048;
static constexpr int BATCH   = 4;
static constexpr int DK      = 64;            // head dim
static constexpr int MTOK    = BATCH * SEQ_T; // 8192 rows

// Toggle for the CDNA5 async-to-LDS DMA path (ASYNCcnt-tracked).
#define USE_ASYNC_LDS 1

__device__ __forceinline__ v8f wmma_bf16(v16bf a, v16bf b, v8f c) {
  return __builtin_amdgcn_wmma_f32_16x16x32_bf16(
      /*neg_a=*/false, a, /*neg_b=*/false, b,
      /*c_mod=*/(short)0, c, /*reuse_a=*/false, /*reuse_b=*/false);
}

// Load one 16-bit A/B operand fragment (ISA 16-bit 16x32 layout):
// lane L -> row = L&15, khalf = (L>>4)*8; elems [0..7]=p[0..7],
// elems [8..15]=p[16..23] with p at (row, khalf) of a K-contiguous row.
__device__ __forceinline__ v16bf ldfrag(const bf16* p) {
  v8bf lo = *reinterpret_cast<const v8bf*>(p);
  v8bf hi = *reinterpret_cast<const v8bf*>(p + 16);
  return __builtin_shufflevector(lo, hi, 0, 1, 2, 3, 4, 5, 6, 7,
                                 8, 9, 10, 11, 12, 13, 14, 15);
}

// Async copy of one 16B chunk global->LDS (per-lane), tracked by ASYNCcnt.
__device__ __forceinline__ void async_cp16(const bf16* g, bf16* l) {
#if USE_ASYNC_LDS
  unsigned long long ga = (unsigned long long)(uintptr_t)g;
  unsigned int la = (unsigned int)(uintptr_t)l;  // AS3 offset = low 32 bits
  asm volatile("global_load_async_to_lds_b128 %0, %1, off"
               :: "v"(la), "v"(ga)
               : "memory");
#else
  *reinterpret_cast<v8bf*>(l) = *reinterpret_cast<const v8bf*>(g);
#endif
}

__device__ __forceinline__ void async_wait0() {
#if USE_ASYNC_LDS
  asm volatile("s_wait_asynccnt 0x0" ::: "memory");
#endif
}

// ---------------------------------------------------------------------------
// f32 -> bf16 conversion
// ---------------------------------------------------------------------------
__global__ __launch_bounds__(256) void cvt_bf16_kernel(
    const float* __restrict__ in, bf16* __restrict__ out, int n) {
  int i = blockIdx.x * 256 + threadIdx.x;
  if (i < n) out[i] = (bf16)in[i];
}

// ---------------------------------------------------------------------------
// GEMM: C[M,N] = act(A[M,K] @ W[N,K]^T + bias[N]),  A/W bf16 row-major.
// Block = 256 thr (8 waves, 4Mx2N).  Block tile 128(M) x 64(N), K-tile 64.
// A/W tiles staged in LDS via async-to-LDS DMA, double buffered; each wave
// computes a 32x32 tile = 2x2 WMMA fragments per 32-K step.
// ---------------------------------------------------------------------------
static constexpr int BM = 128, BN = 64, BK = 64;
static constexpr int LDS_STRIDE = 72;  // 64 + 8 pad (144B rows, 16B aligned)

template <bool RELU, bool OUT_F32, bool OUT_BF16>
__global__ __launch_bounds__(256) void gemm_bf16_kernel(
    const bf16* __restrict__ A, const bf16* __restrict__ W,
    const float* __restrict__ bias, float* __restrict__ Cf,
    bf16* __restrict__ Cb, int M, int N, int K) {
  __shared__ alignas(16) bf16 sA[2][BM * LDS_STRIDE];  // 2 x 18 KB
  __shared__ alignas(16) bf16 sW[2][BN * LDS_STRIDE];  // 2 x  9 KB

  const int tid = threadIdx.x;
  const int lane = tid & 31;
  const int wave = tid >> 5;
  const int m0l = (wave & 3) * 32;   // wave M offset within block tile
  const int n0l = (wave >> 2) * 32;  // wave N offset within block tile

  const int rlane = lane & 15;
  const int koff = (lane >> 4) << 3;  // 0 or 8
  const int half = lane >> 4;

  const bf16* Ab = A + (size_t)blockIdx.x * BM * K;
  const bf16* Wb = W + (size_t)blockIdx.y * BN * K;

  // Stage a [rows x 64] K-slab (row-major, ld=K) into LDS, 16B chunks.
  auto stage = [&](const bf16* g, bf16* l, int rows, int kofs) {
    const int nch = rows * 8;  // 8 x 16B chunks per row
    for (int c = tid; c < nch; c += 256) {
      int r = c >> 3, cc = (c & 7) << 3;
      async_cp16(g + (size_t)r * K + kofs + cc, l + r * LDS_STRIDE + cc);
    }
  };

  v8f acc00 = {}, acc01 = {}, acc10 = {}, acc11 = {};

  stage(Ab, sA[0], BM, 0);
  stage(Wb, sW[0], BN, 0);

  const int nk = K / BK;
  for (int t = 0; t < nk; ++t) {
    async_wait0();
    __syncthreads();
    if (t + 1 < nk) {
      stage(Ab, sA[(t + 1) & 1], BM, (t + 1) * BK);
      stage(Wb, sW[(t + 1) & 1], BN, (t + 1) * BK);
    }
    const bf16* a0 = sA[t & 1] + (m0l + rlane) * LDS_STRIDE + koff;
    const bf16* a1 = a0 + 16 * LDS_STRIDE;
    const bf16* w0 = sW[t & 1] + (n0l + rlane) * LDS_STRIDE + koff;
    const bf16* w1 = w0 + 16 * LDS_STRIDE;
#pragma unroll
    for (int kk = 0; kk < BK; kk += 32) {
      v16bf af0 = ldfrag(a0 + kk);
      v16bf af1 = ldfrag(a1 + kk);
      v16bf wf0 = ldfrag(w0 + kk);
      v16bf wf1 = ldfrag(w1 + kk);
      acc00 = wmma_bf16(af0, wf0, acc00);
      acc01 = wmma_bf16(af0, wf1, acc01);
      acc10 = wmma_bf16(af1, wf0, acc10);
      acc11 = wmma_bf16(af1, wf1, acc11);
    }
    // No trailing barrier: next stage() writes the *other* buffer, and the
    // top-of-loop wait+barrier orders buffer reuse across iterations.
  }

  // Epilogue.  C layout: VGPR r, lane L -> row = r + 8*(L>>4), col = L&15.
  const int m0 = blockIdx.x * BM + m0l;
  const int n0 = blockIdx.y * BN + n0l;
#pragma unroll
  for (int i = 0; i < 2; ++i) {
#pragma unroll
    for (int j = 0; j < 2; ++j) {
      const v8f& acc = (i == 0) ? (j == 0 ? acc00 : acc01)
                                : (j == 0 ? acc10 : acc11);
      int col = n0 + j * 16 + rlane;
      float bv = bias[col];
#pragma unroll
      for (int r = 0; r < 8; ++r) {
        int row = m0 + i * 16 + r + half * 8;
        float v = acc[r] + bv;
        if (RELU) v = fmaxf(v, 0.0f);
        if (OUT_F32) Cf[(size_t)row * N + col] = v;
        if (OUT_BF16) Cb[(size_t)row * N + col] = (bf16)v;
      }
    }
  }
}

// ---------------------------------------------------------------------------
// Transpose V: [B*T, D] (head h cols h*64..h*64+63) -> VT[(b*H+h)*64+d][T]
// ---------------------------------------------------------------------------
__global__ __launch_bounds__(256) void vtrans_kernel(
    const bf16* __restrict__ V, bf16* __restrict__ VT) {
  int idx = blockIdx.x * 256 + threadIdx.x;  // over B*H*DK*T = 8M
  if (idx >= BATCH * N_HEADS * DK * SEQ_T) return;
  int t = idx & (SEQ_T - 1);
  int rest = idx >> 11;
  int d = rest & (DK - 1);
  int bh = rest >> 6;
  int h = bh & (N_HEADS - 1);
  int b = bh >> 4;
  VT[idx] = V[((size_t)(b * SEQ_T + t)) * D_MODEL + h * DK + d];
}

// ---------------------------------------------------------------------------
// Causal flash attention.  Grid: B*H*(T/128) blocks, 256 thr = 8 waves.
// Wave: 16 queries x dk=64, online softmax over 32-key tiles.
// ---------------------------------------------------------------------------
__global__ __launch_bounds__(256) void attn_kernel(
    const bf16* __restrict__ Q, const bf16* __restrict__ Kmat,
    const bf16* __restrict__ VT, bf16* __restrict__ O) {
  const int lane = threadIdx.x & 31;
  const int wave = threadIdx.x >> 5;
  const int blk = blockIdx.x;
  const int qchunk = blk & 15;  // T/128 = 16 chunks
  const int bh = blk >> 4;      // 0..63
  const int b = bh >> 4;
  const int h = bh & (N_HEADS - 1);
  const int q_base = qchunk * 128 + wave * 16;

  const int rlane = lane & 15;
  const int khalf = lane >> 4;
  const int koff = khalf * 8;

  const bf16* qp = Q + (size_t)b * SEQ_T * D_MODEL + h * DK;
  const bf16* kp = Kmat + (size_t)b * SEQ_T * D_MODEL + h * DK;
  const bf16* vtp = VT + (size_t)bh * DK * SEQ_T;

  const bf16* qrow = qp + (size_t)(q_base + rlane) * D_MODEL;
  v16bf qf0 = ldfrag(qrow + koff);
  v16bf qf1 = ldfrag(qrow + 32 + koff);

  float rowM[8], rowL[8];
  v8f acc[4] = {{}, {}, {}, {}};
#pragma unroll
  for (int r = 0; r < 8; ++r) { rowM[r] = -1.0e30f; rowL[r] = 0.0f; }

  __shared__ bf16 pbuf[8][16][40];
  bf16(*pt)[40] = pbuf[wave];

  const float scale = 0.125f;          // 1/sqrt(64)
  const int nkt = (q_base + 47) >> 5;  // # of 32-key tiles (causal)

  for (int kt32 = 0; kt32 < nkt; ++kt32) {
    const int kt = kt32 * 32;
    v8f s0 = {}, s1 = {};
    {
      const bf16* kr0 = kp + (size_t)(kt + rlane) * D_MODEL + koff;
      const bf16* kr1 = kr0 + (size_t)16 * D_MODEL;
      v16bf kf;
      kf = ldfrag(kr0);       s0 = wmma_bf16(qf0, kf, s0);
      kf = ldfrag(kr0 + 32);  s0 = wmma_bf16(qf1, kf, s0);
      kf = ldfrag(kr1);       s1 = wmma_bf16(qf0, kf, s1);
      kf = ldfrag(kr1 + 32);  s1 = wmma_bf16(qf1, kf, s1);
    }
#pragma unroll
    for (int r = 0; r < 8; ++r) {
      const int qi = q_base + r + khalf * 8;
      const float a = (kt + rlane <= qi) ? (float)s0[r] * scale : -1.0e30f;
      const float c = (kt + 16 + rlane <= qi) ? (float)s1[r] * scale : -1.0e30f;
      float m = fmaxf(a, c);
#pragma unroll
      for (int off = 8; off >= 1; off >>= 1)
        m = fmaxf(m, __shfl_xor(m, off, 16));
      const float newM = fmaxf(rowM[r], m);
      const float corr = __expf(rowM[r] - newM);
      rowM[r] = newM;
      const float p0 = __expf(a - newM);
      const float p1 = __expf(c - newM);
      float ps = p0 + p1;
#pragma unroll
      for (int off = 8; off >= 1; off >>= 1) ps += __shfl_xor(ps, off, 16);
      rowL[r] = rowL[r] * corr + ps;
#pragma unroll
      for (int dg = 0; dg < 4; ++dg) acc[dg][r] *= corr;
      pt[r + khalf * 8][rlane] = (bf16)p0;
      pt[r + khalf * 8][16 + rlane] = (bf16)p1;
    }
    v16bf pf = ldfrag(&pt[rlane][koff]);
#pragma unroll
    for (int dg = 0; dg < 4; ++dg) {
      const bf16* vr = vtp + (size_t)(dg * 16 + rlane) * SEQ_T + kt + koff;
      v16bf vf = ldfrag(vr);
      acc[dg] = wmma_bf16(pf, vf, acc[dg]);
    }
  }

#pragma unroll
  for (int r = 0; r < 8; ++r) {
    const float inv = 1.0f / rowL[r];
    const int row = q_base + r + khalf * 8;
    bf16* op = O + ((size_t)(b * SEQ_T + row)) * D_MODEL + h * DK + rlane;
#pragma unroll
    for (int dg = 0; dg < 4; ++dg) op[dg * 16] = (bf16)(acc[dg][r] * inv);
  }
}

// ---------------------------------------------------------------------------
// LayerNorm(residual): out = LN(A + R) * g + be ; optional bf16 copy.
// ---------------------------------------------------------------------------
__global__ __launch_bounds__(256) void ln_residual_kernel(
    const float* __restrict__ A, const float* __restrict__ R,
    const float* __restrict__ g, const float* __restrict__ be,
    float* __restrict__ outF, bf16* __restrict__ outB) {
  const int row = blockIdx.x;
  const float* a = A + (size_t)row * D_MODEL;
  const float* rr = R + (size_t)row * D_MODEL;
  float v[4];
  float s = 0.0f;
#pragma unroll
  for (int i = 0; i < 4; ++i) {
    int idx = threadIdx.x + i * 256;
    v[i] = a[idx] + rr[idx];
    s += v[i];
  }
  __shared__ float red[8];
  const int wv = threadIdx.x >> 5, ln = threadIdx.x & 31;
#pragma unroll
  for (int off = 16; off >= 1; off >>= 1) s += __shfl_xor(s, off, 32);
  if (ln == 0) red[wv] = s;
  __syncthreads();
  float mean = 0.0f;
#pragma unroll
  for (int i = 0; i < 8; ++i) mean += red[i];
  mean *= (1.0f / D_MODEL);
  __syncthreads();
  float vs = 0.0f;
#pragma unroll
  for (int i = 0; i < 4; ++i) {
    float d = v[i] - mean;
    vs += d * d;
  }
#pragma unroll
  for (int off = 16; off >= 1; off >>= 1) vs += __shfl_xor(vs, off, 32);
  if (ln == 0) red[wv] = vs;
  __syncthreads();
  float var = 0.0f;
#pragma unroll
  for (int i = 0; i < 8; ++i) var += red[i];
  var *= (1.0f / D_MODEL);
  const float rstd = rsqrtf(var + 1e-5f);
#pragma unroll
  for (int i = 0; i < 4; ++i) {
    int idx = threadIdx.x + i * 256;
    float y = (v[i] - mean) * rstd * g[idx] + be[idx];
    outF[(size_t)row * D_MODEL + idx] = y;
    if (outB) outB[(size_t)row * D_MODEL + idx] = (bf16)y;
  }
}

// ---------------------------------------------------------------------------
// Host-side orchestration
// ---------------------------------------------------------------------------
extern "C" void kernel_launch(void* const* d_in, const int* in_sizes, int n_in,
                              void* d_out, int out_size, void* d_ws,
                              size_t ws_size, hipStream_t stream) {
  (void)in_sizes; (void)n_in; (void)out_size; (void)ws_size;
  const float* x  = (const float*)d_in[0];
  const float* WQ = (const float*)d_in[1];
  const float* bQ = (const float*)d_in[2];
  const float* WK = (const float*)d_in[3];
  const float* bK = (const float*)d_in[4];
  const float* WV = (const float*)d_in[5];
  const float* bV = (const float*)d_in[6];
  const float* WO = (const float*)d_in[7];
  const float* bO = (const float*)d_in[8];
  const float* W1 = (const float*)d_in[9];
  const float* b1 = (const float*)d_in[10];
  const float* W2 = (const float*)d_in[11];
  const float* b2 = (const float*)d_in[12];
  const float* g1 = (const float*)d_in[13];
  const float* be1 = (const float*)d_in[14];
  const float* g2 = (const float*)d_in[15];
  const float* be2 = (const float*)d_in[16];
  float* out = (float*)d_out;

  char* w = (char*)d_ws;
  auto carve = [&](size_t bytes) {
    char* p = w;
    w += (bytes + 255) & ~(size_t)255;
    return p;
  };
  const size_t MT = (size_t)MTOK;
  bf16* xb  = (bf16*)carve(MT * D_MODEL * 2);
  bf16* wqb = (bf16*)carve((size_t)D_MODEL * D_MODEL * 2);
  bf16* wkb = (bf16*)carve((size_t)D_MODEL * D_MODEL * 2);
  bf16* wvb = (bf16*)carve((size_t)D_MODEL * D_MODEL * 2);
  bf16* wob = (bf16*)carve((size_t)D_MODEL * D_MODEL * 2);
  bf16* w1b = (bf16*)carve((size_t)D_FF * D_MODEL * 2);
  bf16* w2b = (bf16*)carve((size_t)D_MODEL * D_FF * 2);
  bf16* qb  = (bf16*)carve(MT * D_MODEL * 2);
  bf16* kb  = (bf16*)carve(MT * D_MODEL * 2);
  bf16* vb  = (bf16*)carve(MT * D_MODEL * 2);
  bf16* vtb = (bf16*)carve(MT * D_MODEL * 2);
  float* attnf = (float*)carve(MT * D_MODEL * 4);
  float* y1f   = (float*)carve(MT * D_MODEL * 4);
  bf16*  y1b   = (bf16*)carve(MT * D_MODEL * 2);
  bf16*  h1b   = (bf16*)carve(MT * D_FF * 2);
  bf16* aob = vb;     // v dead after vtrans
  float* fff = attnf; // attn_out dead after LN1

  const int nX = MTOK * D_MODEL;
  const int nW = D_MODEL * D_MODEL;
  const int nWF = D_FF * D_MODEL;

  cvt_bf16_kernel<<<(nX + 255) / 256, 256, 0, stream>>>(x, xb, nX);
  cvt_bf16_kernel<<<(nW + 255) / 256, 256, 0, stream>>>(WQ, wqb, nW);
  cvt_bf16_kernel<<<(nW + 255) / 256, 256, 0, stream>>>(WK, wkb, nW);
  cvt_bf16_kernel<<<(nW + 255) / 256, 256, 0, stream>>>(WV, wvb, nW);
  cvt_bf16_kernel<<<(nW + 255) / 256, 256, 0, stream>>>(WO, wob, nW);
  cvt_bf16_kernel<<<(nWF + 255) / 256, 256, 0, stream>>>(W1, w1b, nWF);
  cvt_bf16_kernel<<<(nWF + 255) / 256, 256, 0, stream>>>(W2, w2b, nWF);

  dim3 blk(256);
  dim3 gProj(MTOK / 128, D_MODEL / 64);
  gemm_bf16_kernel<false, false, true><<<gProj, blk, 0, stream>>>(
      xb, wqb, bQ, nullptr, qb, MTOK, D_MODEL, D_MODEL);
  gemm_bf16_kernel<false, false, true><<<gProj, blk, 0, stream>>>(
      xb, wkb, bK, nullptr, kb, MTOK, D_MODEL, D_MODEL);
  gemm_bf16_kernel<false, false, true><<<gProj, blk, 0, stream>>>(
      xb, wvb, bV, nullptr, vb, MTOK, D_MODEL, D_MODEL);

  vtrans_kernel<<<(nX + 255) / 256, 256, 0, stream>>>(vb, vtb);

  attn_kernel<<<BATCH * N_HEADS * (SEQ_T / 128), blk, 0, stream>>>(
      qb, kb, vtb, aob);

  gemm_bf16_kernel<false, true, false><<<gProj, blk, 0, stream>>>(
      aob, wob, bO, attnf, nullptr, MTOK, D_MODEL, D_MODEL);

  ln_residual_kernel<<<MTOK, blk, 0, stream>>>(x, attnf, g1, be1, y1f, y1b);

  dim3 gFF1(MTOK / 128, D_FF / 64);
  gemm_bf16_kernel<true, false, true><<<gFF1, blk, 0, stream>>>(
      y1b, w1b, b1, nullptr, h1b, MTOK, D_FF, D_MODEL);

  gemm_bf16_kernel<false, true, false><<<gProj, blk, 0, stream>>>(
      h1b, w2b, b2, fff, nullptr, MTOK, D_MODEL, D_FF);

  ln_residual_kernel<<<MTOK, blk, 0, stream>>>(y1f, fff, g2, be2, out,
                                               nullptr);
}
